// SENet_72670846648709
// MI455X (gfx1250) — matrix operand order repaired and checked by reading, not compiled, not run on hardware
//
#include <hip/hip_runtime.h>
#include <hip/hip_bf16.h>

typedef __attribute__((ext_vector_type(16))) __bf16   bf16x16;
typedef __attribute__((ext_vector_type(8)))  float    f32x8;
typedef __attribute__((ext_vector_type(4)))  unsigned u32x4;
typedef __attribute__((ext_vector_type(8)))  unsigned u32x8;

#define B_TOT   16384
#define F_NUM   40
#define D_DIM   64
#define ROW     2560          // F*D
#define RED     53
#define KZ      160           // 2*F*G
#define EPS_LN  1e-3f
#define TILE_ROWS 16

#define ZS   168              // padded Z row stride (bf16 elems, 16B-aligned rows)
#define A1S  72               // padded A1 row stride

// workspace layout (bf16 elements)
#define W2F_ELEMS (160*2*32*16)   // [tile][kstep][lane][16] = 163840 bf16
#define W1F_ELEMS (4*5*32*16)     // [tile][kstep][lane][16] = 10240 bf16

// ---------------- weight pre-swizzle into WMMA B-fragment layout ----------------
// B frag (16x16x32 bf16) as packed here (contiguous k, same order as the A loads): lane = {N=lane&15, Khalf=lane>>4}, slot j -> K = kstep*32 + (lane>>4)*16 + j
__global__ void prep_w2(const float* __restrict__ W2, __bf16* __restrict__ w2f) {
    int idx = blockIdx.x * 256 + threadIdx.x;
    if (idx >= W2F_ELEMS) return;
    int j    = idx & 15;
    int lane = (idx >> 4) & 31;
    int ks   = (idx >> 9) & 1;
    int tile = idx >> 10;
    int k   = ks * 32 + ((lane >> 4) * 16) + j;      // padded K: 0..63
    int col = tile * 16 + (lane & 15);               // 0..2559
    float v = (k < RED) ? W2[k * ROW + col] : 0.0f;  // zero-pad K rows 53..63
    w2f[idx] = (__bf16)v;
}

__global__ void prep_w1(const float* __restrict__ W1, __bf16* __restrict__ w1f) {
    int idx = blockIdx.x * 256 + threadIdx.x;
    if (idx >= W1F_ELEMS) return;
    int j    = idx & 15;
    int lane = (idx >> 4) & 31;
    int rest = idx >> 9;
    int ks   = rest % 5;
    int tile = rest / 5;
    int k   = ks * 32 + ((lane >> 4) * 16) + j;      // 0..159 (exact)
    int col = tile * 16 + (lane & 15);               // padded N: 0..63
    float v = (col < RED) ? W1[k * RED + col] : 0.0f;
    w1f[idx] = (__bf16)v;
}

// ---------------- fused SENet kernel ----------------
union FragBF { uint4 q[2]; bf16x16 v; };

__launch_bounds__(256)
__global__ void senet_fused(const float* __restrict__ x,
                            const float* __restrict__ b1,
                            const float* __restrict__ b2,
                            const float* __restrict__ gmm,
                            const float* __restrict__ bta,
                            const __bf16* __restrict__ w1f,
                            const __bf16* __restrict__ w2f,
                            float* __restrict__ out) {
    extern __shared__ char smem[];
    float*  xT  = (float*)smem;                                        // 16x2560 f32
    __bf16* zT  = (__bf16*)(smem + TILE_ROWS * ROW * 4);               // 16xZS bf16
    __bf16* a1T = (__bf16*)(smem + TILE_ROWS * ROW * 4 + TILE_ROWS * ZS * 2);

    const int tid  = threadIdx.x;
    const int lane = tid & 31;
    const int wv   = tid >> 5;
    const int hi   = lane >> 4;
    const int l15  = lane & 15;
    const long r0  = (long)blockIdx.x * TILE_ROWS;

    // ---- Phase 0: TDM async bulk copy of the 160KB x-tile (contiguous 1-D) into LDS ----
    if (tid < 32) {  // wave 0 issues the tensor DMA (scalar op, descriptor in SGPRs)
        unsigned long long ga = (unsigned long long)(const void*)(x + r0 * ROW);
        unsigned ldsOff = (unsigned)(unsigned long long)(void*)xT;  // addr[31:0] = LDS byte offset
        const unsigned n0 = TILE_ROWS * ROW;                        // 40960 f32 elements
        u32x4 g0;
        g0[0] = 1u;                                                 // count=1, user load desc
        g0[1] = ldsOff;                                             // lds_addr
        g0[2] = (unsigned)ga;                                       // global_addr[31:0]
        g0[3] = ((unsigned)(ga >> 32) & 0x01FFFFFFu) | (2u << 30);  // global_addr[56:32] | type=2
        u32x8 g1;
        g1[0] = 0x00020000u;                 // workgroup_mask=0, data_size=2 (4B)
        g1[1] = (n0 & 0xFFFFu) << 16;        // tensor_dim0[15:0]
        g1[2] = (n0 >> 16) | (1u << 16);     // tensor_dim0[31:16] | tensor_dim1=1
        g1[3] = (n0 & 0xFFFFu) << 16;        // tile_dim0 = 40960
        g1[4] = 1u;                          // tile_dim1 = 1, tile_dim2 = 0
        g1[5] = n0;                          // tensor_dim0_stride[31:0]
        g1[6] = 0u;
        g1[7] = 0u;
        asm volatile("tensor_load_to_lds %0, %1\n\t"
                     "s_wait_tensorcnt 0x0"
                     :: "s"(g0), "s"(g1) : "memory");
    }
    __syncthreads();

    // ---- Phase 1: group pooling -> Z[16,160] bf16 (means at [0:80], maxes at [80:160]) ----
    for (int rr = 0; rr < 2; ++rr) {
        const int r = wv * 2 + rr;
        for (int q = 0; q < 20; ++q) {
            float4 v4 = *(const float4*)&xT[r * ROW + q * 128 + lane * 4];
            float s = v4.x + v4.y + v4.z + v4.w;
            float m = fmaxf(fmaxf(v4.x, v4.y), fmaxf(v4.z, v4.w));
            for (int o = 1; o < 8; o <<= 1) {           // reduce 8-lane groups (32 elems)
                s += __shfl_xor(s, o, 32);
                m = fmaxf(m, __shfl_xor(m, o, 32));
            }
            if ((lane & 7) == 0) {
                int fg = q * 4 + (lane >> 3);
                zT[r * ZS + fg]       = (__bf16)(s * (1.0f / 32.0f));
                zT[r * ZS + 80 + fg]  = (__bf16)m;
            }
        }
    }
    __syncthreads();

    // ---- Phase 2: GEMM1  A1 = relu(Z @ W1 + b1)   [16,160]x[160,64pad] ----
    if (wv < 4) {
        const int n = wv;                    // N-tile
        f32x8 acc = {};
        for (int ks = 0; ks < 5; ++ks) {
            FragBF a, b;
            const __bf16* zp = &zT[l15 * ZS + ks * 32 + hi * 16];
            a.q[0] = *(const uint4*)(zp);
            a.q[1] = *(const uint4*)(zp + 8);
            const uint4* wp = (const uint4*)(w1f + (((n * 5 + ks) * 32 + lane) << 4));
            b.q[0] = wp[0]; b.q[1] = wp[1];
            acc = __builtin_amdgcn_wmma_f32_16x16x32_bf16(false, a.v, false, b.v,
                                                          (short)0, acc, false, false);
        }
        const int col = n * 16 + l15;
        const float bias = (col < RED) ? b1[col] : 0.0f;
        for (int i = 0; i < 8; ++i) {
            float t = fmaxf(acc[i] + bias, 0.0f);
            a1T[(i + hi * 8) * A1S + col] = (__bf16)t;
        }
    }
    __syncthreads();

    // ---- Phase 3: GEMM2 + fused excitation/residual/LayerNorm epilogue ----
    FragBF a1frag[2];
    for (int ks = 0; ks < 2; ++ks) {
        const __bf16* p = &a1T[l15 * A1S + ks * 32 + hi * 16];
        a1frag[ks].q[0] = *(const uint4*)p;
        a1frag[ks].q[1] = *(const uint4*)(p + 8);
    }

    for (int ff = 0; ff < 5; ++ff) {
        const int f = wv * 5 + ff;           // field owned by this wave
        float vv[4][8];
        float gT[4], bT[4];
        float ssum[8], ssq[8];
        for (int i = 0; i < 8; ++i) { ssum[i] = 0.0f; ssq[i] = 0.0f; }

        for (int T = 0; T < 4; ++T) {
            const int tile = f * 4 + T;
            f32x8 acc = {};
            for (int ks = 0; ks < 2; ++ks) {
                FragBF b;
                const uint4* wp = (const uint4*)(w2f + (((tile * 2 + ks) * 32 + lane) << 4));
                b.q[0] = wp[0]; b.q[1] = wp[1];
                acc = __builtin_amdgcn_wmma_f32_16x16x32_bf16(false, a1frag[ks].v, false, b.v,
                                                              (short)0, acc, false, false);
            }
            const int col = tile * 16 + l15;             // = f*64 + T*16 + l15
            const float bias = b2[col];
            gT[T] = gmm[col];
            bT[T] = bta[col];
            for (int i = 0; i < 8; ++i) {
                float w  = fmaxf(acc[i] + bias, 0.0f);   // relu(A1@W2 + b2)
                float xv = xT[(i + hi * 8) * ROW + col];
                float v  = fmaf(xv, w, xv);              // x*w + x
                vv[T][i] = v;
                ssum[i] += v;
                ssq[i]  += v * v;
            }
        }
        // LayerNorm over D=64: 16-lane cross-lane reduction per row (lane-half = row group)
        for (int i = 0; i < 8; ++i) {
            float s = ssum[i], q = ssq[i];
            for (int o = 1; o < 16; o <<= 1) {
                s += __shfl_xor(s, o, 32);
                q += __shfl_xor(q, o, 32);
            }
            const float mu  = s * (1.0f / 64.0f);
            const float var = q * (1.0f / 64.0f) - mu * mu;
            const float rs  = rsqrtf(var + EPS_LN);
            const long orow = (r0 + i + hi * 8) * (long)ROW;
            for (int T = 0; T < 4; ++T) {
                float y = (vv[T][i] - mu) * rs * gT[T] + bT[T];
                out[orow + f * 64 + T * 16 + l15] = y;
            }
        }
    }
}

// ---------------- launcher ----------------
extern "C" void kernel_launch(void* const* d_in, const int* in_sizes, int n_in,
                              void* d_out, int out_size, void* d_ws, size_t ws_size,
                              hipStream_t stream) {
    (void)in_sizes; (void)n_in; (void)out_size; (void)ws_size;
    const float* x   = (const float*)d_in[0];
    const float* W1  = (const float*)d_in[1];
    const float* b1  = (const float*)d_in[2];
    const float* W2  = (const float*)d_in[3];
    const float* b2  = (const float*)d_in[4];
    const float* gmm = (const float*)d_in[5];
    const float* bta = (const float*)d_in[6];
    float* out = (float*)d_out;

    __bf16* w2f = (__bf16*)d_ws;                 // 327680 B
    __bf16* w1f = w2f + W2F_ELEMS;               // +20480 B

    prep_w2<<<(W2F_ELEMS + 255) / 256, 256, 0, stream>>>(W2, w2f);
    prep_w1<<<(W1F_ELEMS + 255) / 256, 256, 0, stream>>>(W1, w1f);

    const size_t shmem = (size_t)TILE_ROWS * ROW * 4
                       + (size_t)TILE_ROWS * ZS * 2
                       + (size_t)TILE_ROWS * A1S * 2;   // 171520 B (<320KB/WGP)
    (void)hipFuncSetAttribute((const void*)senet_fused,
                              hipFuncAttributeMaxDynamicSharedMemorySize, (int)shmem);
    senet_fused<<<B_TOT / TILE_ROWS, 256, shmem, stream>>>(x, b1, b2, gmm, bta, w1f, w2f, out);
}
